// LawinHead_2439541424534
// MI455X (gfx1250) — matrix operand order repaired and checked by reading, not compiled, not run on hardware
//
#include <hip/hip_runtime.h>

// ---------------------------------------------------------------------------
// LawinHead fused attention for MI455X (gfx1250, wave32, WMMA bf16).
// n=1024, c=512, N=64 tokens, HEAD=16, inter=256, dh=16. ~77 GFLOP, ~400 MB.
// All GEMMs via v_wmma_f32_16x16x32_bf16, b128 fragment loads, residual/bias
// folded into the WMMA C operand, b128 packed epilogue stores.
// ---------------------------------------------------------------------------

typedef __bf16 v16bf __attribute__((ext_vector_type(16)));
typedef __bf16 v2bf  __attribute__((ext_vector_type(2)));
typedef float  v8f   __attribute__((ext_vector_type(8)));

#define C_DIM   512
#define N_TOK   64
#define INTER   256
#define NHEAD   16
#define BLOCK   256

// dynamic LDS layout (bytes), total 288 KB of the 320 KB WGP budget
#define OFF_CTX    0        // ctx  row-major [512][64]   (A: stage A2) -> scr
#define OFF_CTXT   65536    // ctxT K-major  [64][512]    (B: stage B)  -> yT
#define OFF_QT     131072   // qT   K-major  [64][512]    (B: stage B theta)
#define OFF_THT    196608   // thetaT [64][256]           (A: scores)
#define OFF_PHT    229376   // phiT   [64][256]           (B: scores)
#define OFF_G      262144   // g row-major [256][64]      (B: y, K=token)
#define SMEM_BYTES 294912

// d_ws layout (bf16 element offsets) -- straight bf16 copies of the weights
#define WS_WM  0            // [16][64][64]  Wm[h][o][k] (K-contig per o col)
#define WS_WG  65536        // [256][512]
#define WS_WT  196608       // [256][512]
#define WS_WP  327680       // [256][512]
#define WS_WO  458752       // [512][256]

__device__ __forceinline__ unsigned cvt2(float a, float b) {
  union { v2bf v; unsigned u; } x;
  x.v[0] = (__bf16)a;                    // native v_cvt (RTNE)
  x.v[1] = (__bf16)b;
  return x.u;
}
__device__ __forceinline__ uint4 cvt8(v8f s) {
  uint4 r;
  r.x = cvt2(s[0], s[1]); r.y = cvt2(s[2], s[3]);
  r.z = cvt2(s[4], s[5]); r.w = cvt2(s[6], s[7]);
  return r;
}
__device__ __forceinline__ v8f zero_v8f() {
  v8f z;
#pragma unroll
  for (int i = 0; i < 8; ++i) z[i] = 0.0f;
  return z;
}
__device__ __forceinline__ v8f wmma_bf16(v16bf a, v16bf b, v8f c) {
  return __builtin_amdgcn_wmma_f32_16x16x32_bf16(false, a, false, b,
                                                 (short)0, c, false, false);
}

union Frag { v16bf v; uint4 q[2]; };

// A fragment 16x32 (MxK), source row-major base[(r0+m)*stride + k].
// Lane layout: m=l&15; kb=(l<16)?0:8; dwords 0-3 hold K=kb..kb+7,
// dwords 4-7 hold K=kb+16..kb+23  -> two contiguous 16B runs.
__device__ __forceinline__ v16bf loadA(const __bf16* base, int r0,
                                       int k0, int stride) {
  const int l = threadIdx.x & 31;
  const __bf16* p = base + (r0 + (l & 15)) * stride + k0 + ((l < 16) ? 0 : 8);
  Frag f;
  f.q[0] = *(const uint4*)p;
  f.q[1] = *(const uint4*)(p + 16);
  return f.v;
}

// A fragment for scores: A[q,d] = thetaT[(c0+q)*256 + r0 + d], K=16 pad to 32.
__device__ __forceinline__ v16bf loadA_thetaT(const __bf16* base,
                                              int r0, int c0) {
  const int l = threadIdx.x & 31;
  const __bf16* p = base + (c0 + (l & 15)) * INTER + r0 + ((l < 16) ? 0 : 8);
  Frag f;
  f.q[0] = *(const uint4*)p;             // K = kb .. kb+7 (all real, d<16)
  f.q[1] = make_uint4(0, 0, 0, 0);       // K = 16..31 zero-padded
  return f.v;
}

// B fragment 32x16 (KxN), source K-contiguous per column:
// element (k, n) at base[(c0+n)*stride + k].
// Lane layout: n=l&15; kb=(l<16)?0:8; dwords 0-3 hold K=kb..kb+7,
// dwords 4-7 hold K=kb+16..kb+23 -> two contiguous 16B runs.
__device__ __forceinline__ v16bf loadB_kc(const __bf16* base, int k0,
                                          int c0, int stride) {
  const int l = threadIdx.x & 31;
  const __bf16* p = base + (c0 + (l & 15)) * stride + k0 + ((l < 16) ? 0 : 8);
  Frag f;
  f.q[0] = *(const uint4*)p;
  f.q[1] = *(const uint4*)(p + 16);
  return f.v;
}

// ---------------------------------------------------------------------------
// Prepass: all weights fp32 -> bf16 straight copies into d_ws (pairwise).
// ---------------------------------------------------------------------------
__global__ void lawin_prep(const float* __restrict__ Wm,
                           const float* __restrict__ Wg,
                           const float* __restrict__ Wt,
                           const float* __restrict__ Wp,
                           const float* __restrict__ Wo,
                           unsigned* __restrict__ ws32) {
  const int idx = blockIdx.x * blockDim.x + threadIdx.x;
  const int stride = gridDim.x * blockDim.x;
  const float2* m2 = (const float2*)Wm;
  const float2* g2 = (const float2*)Wg;
  const float2* t2 = (const float2*)Wt;
  const float2* p2 = (const float2*)Wp;
  const float2* o2 = (const float2*)Wo;
  for (int i = idx; i < (NHEAD * 64 * 64) / 2; i += stride) {
    float2 v = m2[i];
    ws32[(WS_WM >> 1) + i] = cvt2(v.x, v.y);
  }
  for (int i = idx; i < (INTER * C_DIM) / 2; i += stride) {
    float2 a = g2[i], b = t2[i], c = p2[i], d = o2[i];
    ws32[(WS_WG >> 1) + i] = cvt2(a.x, a.y);
    ws32[(WS_WT >> 1) + i] = cvt2(b.x, b.y);
    ws32[(WS_WP >> 1) + i] = cvt2(c.x, c.y);
    ws32[(WS_WO >> 1) + i] = cvt2(d.x, d.y);
  }
}

// ---------------------------------------------------------------------------
// Main fused kernel: one workgroup (8 wave32) per batch element.
// ---------------------------------------------------------------------------
__global__ void __launch_bounds__(BLOCK)
lawin_main(const float* __restrict__ query, const float* __restrict__ context,
           const float* __restrict__ bm, const float* __restrict__ bg,
           const float* __restrict__ bt, const float* __restrict__ bp,
           const float* __restrict__ bo, const __bf16* __restrict__ ws,
           float* __restrict__ out) {
  extern __shared__ unsigned char smem[];
  __bf16* sCtx  = (__bf16*)(smem + OFF_CTX);   // [512][64] row-major
  __bf16* sCtxT = (__bf16*)(smem + OFF_CTXT);  // [64][512] K-major
  __bf16* sYT   = (__bf16*)(smem + OFF_CTXT);  // [64][256] reuse
  __bf16* sQT   = (__bf16*)(smem + OFF_QT);    // [64][512] K-major
  __bf16* sThT  = (__bf16*)(smem + OFF_THT);   // [64][256]
  __bf16* sPhT  = (__bf16*)(smem + OFF_PHT);   // [64][256]
  __bf16* sG    = (__bf16*)(smem + OFF_G);     // [256][64] row-major

  const int n    = blockIdx.x;
  const int tid  = threadIdx.x;
  const int wave = tid >> 5;
  const int lane = tid & 31;
  const int hf   = (lane >> 4) << 3;   // C/D row offset for upper lane half
  const int ln   = lane & 15;          // C/D column within tile

  const float* cgf = context + (size_t)n * (C_DIM * N_TOK);
  const float* qgf = query + (size_t)n * (C_DIM * N_TOK);

  // ---- Stage A: ctx -> row-major bf16; query -> K-major bf16 --------------
  {
    const float4* cg4 = (const float4*)cgf;
    const float4* qg4 = (const float4*)qgf;
    for (int i = tid; i < (C_DIM * N_TOK) / 4; i += BLOCK) {
      float4 cv = cg4[i], qv = qg4[i];
      const int b = i * 4, c = b >> 6, t = b & 63;
      uint2 pk;
      pk.x = cvt2(cv.x, cv.y);
      pk.y = cvt2(cv.z, cv.w);
      *(uint2*)(sCtx + b) = pk;                 // row-major, vectorized
      sQT[(t + 0) * C_DIM + c] = (__bf16)qv.x;  // K-major scatter
      sQT[(t + 1) * C_DIM + c] = (__bf16)qv.y;
      sQT[(t + 2) * C_DIM + c] = (__bf16)qv.z;
      sQT[(t + 3) * C_DIM + c] = (__bf16)qv.w;
    }
  }
  __syncthreads();

  // ---- Stage A2: ctxT = T(ctx + ctx_r.Wm^T + bm)  (per-head token mixing) -
  // Residual + bias preloaded into the WMMA C operand (fp32 ctx from L2).
  for (int t = wave; t < 32; t += 8) {
    const int r0 = t << 4;
    const int h  = t >> 1;
    const __bf16* wm = ws + WS_WM + h * 4096;    // [o][k] K-contig
    v8f acc[4];
#pragma unroll
    for (int nt = 0; nt < 4; ++nt) {
      const int col = nt * 16 + ln;
      const float bmv = bm[h * 64 + col];
      v8f c;
#pragma unroll
      for (int v = 0; v < 8; ++v)
        c[v] = cgf[(r0 + v + hf) * N_TOK + col] + bmv;
      acc[nt] = c;
    }
#pragma unroll
    for (int ks = 0; ks < 2; ++ks) {
      v16bf a = loadA(sCtx, r0, ks * 32, N_TOK);       // 2x ds b128
#pragma unroll
      for (int nt = 0; nt < 4; ++nt) {
        v16bf b = loadB_kc(wm, ks * 32, nt * 16, 64);  // 2x global b128
        acc[nt] = wmma_bf16(a, b, acc[nt]);
      }
    }
#pragma unroll
    for (int nt = 0; nt < 4; ++nt) {
      const int col = nt * 16 + ln;
      // rows r0+hf..r0+hf+7 are consecutive in the K-major layout
      *(uint4*)(sCtxT + col * C_DIM + r0 + hf) = cvt8(acc[nt]);
    }
  }
  __syncthreads();

  // ---- Stage B: g = Wg.ctx, phi = Wp.ctx, theta = Wt.query ----------------
  for (int j = wave; j < 48; j += 8) {
    const int p  = j >> 4;
    const int m0 = (j & 15) << 4;
    const __bf16* W;
    const __bf16* src;
    const float* bias;
    if (p == 0)      { W = ws + WS_WG; src = sCtxT; bias = bg; }
    else if (p == 1) { W = ws + WS_WP; src = sCtxT; bias = bp; }
    else             { W = ws + WS_WT; src = sQT;   bias = bt; }
    v8f bc;                                      // bias into C operand
#pragma unroll
    for (int v = 0; v < 8; ++v) bc[v] = bias[m0 + v + hf];
    v8f acc[4];
#pragma unroll
    for (int nt = 0; nt < 4; ++nt) acc[nt] = bc;
    for (int ks = 0; ks < 16; ++ks) {
      v16bf a = loadA(W, m0, ks * 32, C_DIM);          // 2x global b128
      if (ks < 15) __builtin_prefetch(W + m0 * C_DIM + (ks + 1) * 32, 0, 1);
#pragma unroll
      for (int nt = 0; nt < 4; ++nt) {
        v16bf b = loadB_kc(src, ks * 32, nt * 16, C_DIM);  // 2x ds b128
        acc[nt] = wmma_bf16(a, b, acc[nt]);
      }
    }
#pragma unroll
    for (int nt = 0; nt < 4; ++nt) {
      const int col = nt * 16 + ln;
      if (p == 0) {                                // g row-major: scatter b16
#pragma unroll
        for (int v = 0; v < 8; ++v)
          sG[(m0 + v + hf) * N_TOK + col] = (__bf16)acc[nt][v];
      } else {                                     // K-major: packed b128
        __bf16* dst = (p == 1) ? sPhT : sThT;
        *(uint4*)(dst + col * INTER + m0 + hf) = cvt8(acc[nt]);
      }
    }
  }
  __syncthreads();

  // ---- Stage C: per-head attention (one wave per head) --------------------
  __bf16* scr = ((__bf16*)(smem + OFF_CTX)) + wave * 4096;   // 64x64 per wave
  for (int h = wave; h < NHEAD; h += 8) {
    const int hr0 = h << 4;
    // scores = theta^T.phi / sqrt(dh); K=16 padded to 32 (A zeros mask B)
    v16bf bP[4];
#pragma unroll
    for (int nt = 0; nt < 4; ++nt)
      bP[nt] = loadB_kc(sPhT + hr0, 0, nt * 16, INTER);
    v8f S[4][4];
#pragma unroll
    for (int mt = 0; mt < 4; ++mt) {
      v16bf a = loadA_thetaT(sThT, hr0, mt * 16);
#pragma unroll
      for (int nt = 0; nt < 4; ++nt)
        S[mt][nt] = wmma_bf16(a, bP[nt], zero_v8f());
    }
    // softmax over k: each row lives in one 16-lane half x 4 n-tiles
#pragma unroll
    for (int mt = 0; mt < 4; ++mt) {
#pragma unroll
      for (int v = 0; v < 8; ++v) {
        float mx = -3.4e38f;
#pragma unroll
        for (int nt = 0; nt < 4; ++nt) {
          S[mt][nt][v] *= 0.25f;                       // 1/sqrt(16)
          mx = fmaxf(mx, S[mt][nt][v]);
        }
        for (int d = 1; d < 16; d <<= 1) mx = fmaxf(mx, __shfl_xor(mx, d, 32));
        float sum = 0.0f;
#pragma unroll
        for (int nt = 0; nt < 4; ++nt) {
          float e = __expf(S[mt][nt][v] - mx);
          S[mt][nt][v] = e;
          sum += e;
        }
        for (int d = 1; d < 16; d <<= 1) sum += __shfl_xor(sum, d, 32);
        const float inv = 1.0f / sum;
#pragma unroll
        for (int nt = 0; nt < 4; ++nt) S[mt][nt][v] *= inv;
      }
    }
    // bounce w through per-wave LDS scratch (D-layout -> A-layout)
#pragma unroll
    for (int mt = 0; mt < 4; ++mt)
#pragma unroll
      for (int nt = 0; nt < 4; ++nt)
#pragma unroll
        for (int v = 0; v < 8; ++v)
          scr[(mt * 16 + v + hf) * 64 + nt * 16 + ln] = (__bf16)S[mt][nt][v];
    asm volatile("s_wait_dscnt 0x0" ::: "memory");     // stores before reloads
    // y^T = w.g^T  (M=q, N=d=16, K=64); g row-major is K-contig per d column
    v16bf bG[2];
#pragma unroll
    for (int ks = 0; ks < 2; ++ks)
      bG[ks] = loadB_kc(sG, ks * 32, hr0, N_TOK);
#pragma unroll
    for (int mt = 0; mt < 4; ++mt) {
      v8f Y = zero_v8f();
#pragma unroll
      for (int ks = 0; ks < 2; ++ks) {
        v16bf a = loadA(scr, mt * 16, ks * 32, 64);
        Y = wmma_bf16(a, bG[ks], Y);
      }
#pragma unroll
      for (int v = 0; v < 8; ++v)                      // yT K-major scatter
        sYT[(mt * 16 + v + hf) * INTER + hr0 + ln] = (__bf16)Y[v];
    }
  }
  __syncthreads();

  // ---- Stage D: out = Wo.y + bo + query (bias+residual in C operand) ------
  float* og = out + (size_t)n * (C_DIM * N_TOK);
  for (int mt = wave; mt < 32; mt += 8) {
    const int m0 = mt << 4;
    v8f acc[4];
#pragma unroll
    for (int nt = 0; nt < 4; ++nt) {
      const int col = nt * 16 + ln;
      v8f c;
#pragma unroll
      for (int v = 0; v < 8; ++v)
        c[v] = bo[m0 + v + hf] + qgf[(m0 + v + hf) * N_TOK + col];
      acc[nt] = c;
    }
    for (int ks = 0; ks < 8; ++ks) {
      v16bf a = loadA(ws + WS_WO, m0, ks * 32, INTER);
      if (ks < 7) __builtin_prefetch(ws + WS_WO + m0 * INTER + (ks + 1) * 32, 0, 1);
#pragma unroll
      for (int nt = 0; nt < 4; ++nt) {
        v16bf b = loadB_kc(sYT, ks * 32, nt * 16, INTER);
        acc[nt] = wmma_bf16(a, b, acc[nt]);
      }
    }
#pragma unroll
    for (int nt = 0; nt < 4; ++nt) {
      const int col = nt * 16 + ln;
#pragma unroll
      for (int v = 0; v < 8; ++v)
        og[(m0 + v + hf) * N_TOK + col] = acc[nt][v];
    }
  }
}

// ---------------------------------------------------------------------------
extern "C" void kernel_launch(void* const* d_in, const int* in_sizes, int n_in,
                              void* d_out, int out_size, void* d_ws,
                              size_t ws_size, hipStream_t stream) {
  (void)in_sizes; (void)n_in; (void)out_size; (void)ws_size;
  const float* query   = (const float*)d_in[0];
  const float* context = (const float*)d_in[1];
  const float* Wm = (const float*)d_in[2];
  const float* bm = (const float*)d_in[3];
  const float* Wg = (const float*)d_in[4];
  const float* bg = (const float*)d_in[5];
  const float* Wt = (const float*)d_in[6];
  const float* bt = (const float*)d_in[7];
  const float* Wp = (const float*)d_in[8];
  const float* bp = (const float*)d_in[9];
  const float* Wo = (const float*)d_in[10];
  const float* bo = (const float*)d_in[11];
  float* out = (float*)d_out;

  lawin_prep<<<256, 256, 0, stream>>>(Wm, Wg, Wt, Wp, Wo, (unsigned*)d_ws);

  (void)hipFuncSetAttribute((const void*)lawin_main,
                            hipFuncAttributeMaxDynamicSharedMemorySize,
                            SMEM_BYTES);
  lawin_main<<<1024, BLOCK, SMEM_BYTES, stream>>>(query, context, bm, bg, bt,
                                                  bp, bo, (const __bf16*)d_ws,
                                                  out);
}
